// SMambaBlock_4569845202961
// MI455X (gfx1250) — matrix operand order repaired and checked by reading, not compiled, not run on hardware
//
#include <hip/hip_runtime.h>
#include <hip/hip_bf16.h>
#include <math.h>

// ---------------- problem constants (from reference) ----------------
#define B_      8
#define L_      2048
#define DMODEL  512
#define DINNER  1024
#define DSTATE  16
#define DTRANK  32
#define DFF     2048
#define M_TOK   (B_ * L_)        // 16384 tokens, flattened (b,l)

typedef unsigned short u16;
typedef __bf16 bf16_t;
typedef float v8f   __attribute__((ext_vector_type(8)));
typedef bf16_t v16bf __attribute__((ext_vector_type(16)));
typedef unsigned short u16x8  __attribute__((ext_vector_type(8)));
typedef unsigned short u16x16 __attribute__((ext_vector_type(16)));
typedef unsigned int   u32x4  __attribute__((ext_vector_type(4)));
typedef unsigned int   uint32x4 __attribute__((ext_vector_type(4)));
typedef int            int32x4  __attribute__((ext_vector_type(4)));
typedef int            int32x8  __attribute__((ext_vector_type(8)));

#if __has_builtin(__builtin_amdgcn_tensor_load_to_lds)
#define HAVE_TDM 1
#else
#define HAVE_TDM 0
#endif

__device__ __forceinline__ u16 f2bf(float f) {
  unsigned u = __float_as_uint(f);
  u += 0x7FFFu + ((u >> 16) & 1u);          // round-to-nearest-even
  return (u16)(u >> 16);
}

// ---------------- elementwise cast kernels ----------------
__global__ void k_cast_bf16(const float* __restrict__ s, u16* __restrict__ d, int n) {
  int i = blockIdx.x * blockDim.x + threadIdx.x;
  if (i < n) d[i] = f2bf(s[i]);
}

// flip along L while casting (for the backward-direction Mamba)
__global__ void k_cast_bf16_flipL(const float* __restrict__ s, u16* __restrict__ d,
                                  int Ln, int Dn, int n) {
  int i = blockIdx.x * blockDim.x + threadIdx.x;
  if (i >= n) return;
  int c  = i % Dn;
  int bl = i / Dn;
  int l  = bl % Ln;
  int b  = bl / Ln;
  d[((size_t)(b * Ln + (Ln - 1 - l)) * Dn) + c] = f2bf(s[i]);
}

// ---------------- TDM: 2D tile (rows x BK bf16) global -> LDS ----------------
#if HAVE_TDM
__device__ __forceinline__ void tdm_load_2d(unsigned lds_byte_addr, const void* gptr,
                                            unsigned tile_k, unsigned tile_rows,
                                            unsigned tensor_k, unsigned tensor_rows,
                                            unsigned long long row_stride_elems) {
  unsigned long long ga = (unsigned long long)(uintptr_t)gptr;
  uint32x4 g0;
  g0[0] = 1u;                                                  // count=1, user descriptor
  g0[1] = lds_byte_addr;                                       // lds_addr [63:32]
  g0[2] = (unsigned)(ga & 0xFFFFFFFFu);                        // global_addr low
  g0[3] = (unsigned)((ga >> 32) & 0x01FFFFFFu) | 0x80000000u;  // addr hi | type=2
  // group1: mask=0, data_size=1 (2B), dims/strides
  unsigned w0 = 0x00010000u;
  unsigned w1 = (tensor_k & 0xFFFFu) << 16;                    // tensor_dim0[15:0] @48
  unsigned w2 = ((tensor_k >> 16) & 0xFFFFu) | ((tensor_rows & 0xFFFFu) << 16);
  unsigned w3 = ((tensor_rows >> 16) & 0xFFFFu) | (tile_k << 16);   // tile_dim0 @112
  unsigned w4 = tile_rows & 0xFFFFu;                           // tile_dim1 @128, tile_dim2=0
  unsigned long long s0 = row_stride_elems;                    // tensor_dim0_stride @160 (48b)
  unsigned w5 = (unsigned)(s0 & 0xFFFFFFFFull);
  unsigned w6 = (unsigned)((s0 >> 32) & 0xFFFFull);            // dim1_stride=0
  unsigned w7 = 0u;
  int32x8 g1;
  g1[0] = (int)w0; g1[1] = (int)w1; g1[2] = (int)w2; g1[3] = (int)w3;
  g1[4] = (int)w4; g1[5] = (int)w5; g1[6] = (int)w6; g1[7] = (int)w7;
  int32x4 gz = {0, 0, 0, 0};
#if defined(__clang_major__) && (__clang_major__ >= 23)
  int32x8 gz8 = {0, 0, 0, 0, 0, 0, 0, 0};
  __builtin_amdgcn_tensor_load_to_lds(g0, g1, gz, gz, gz8, 0);
#else
  __builtin_amdgcn_tensor_load_to_lds(g0, g1, gz, gz, 0);
#endif
}
#endif

// ---------------- WMMA GEMM: C(M,N) = act(A(MxK,bf16) * W(NxK,bf16)^T + bias) ----------------
#define BM 256
#define BN 64
#define BK 32
enum { ACT_NONE = 0, ACT_GELU = 2, ACT_SOFTPLUS = 3 };

__device__ __forceinline__ float apply_act(float x, int act) {
  if (act == ACT_GELU)     return 0.5f * x * (1.f + erff(x * 0.70710678118f));
  if (act == ACT_SOFTPLUS) return (x > 20.f) ? x : log1pf(__expf(x));
  return x;
}

__launch_bounds__(256, 2)
__global__ void k_gemm_bf16_wmma(const u16* __restrict__ A, int lda,
                                 const u16* __restrict__ W, int ldw,
                                 float* __restrict__ C, u16* __restrict__ Cbf,
                                 const float* __restrict__ bias,
                                 int N, int K, int act) {
  const int tid   = threadIdx.x;
  const int lane  = tid & 31;
  const int wid   = tid >> 5;                 // 8 waves
  const int waveM = wid & 3;                  // 4 along M (64 rows each)
  const int waveN = wid >> 2;                 // 2 along N (32 cols each)
  const int m0    = blockIdx.y * BM;
  const int n0    = blockIdx.x * BN;

  v8f acc[4][2] = {};
  const int nk = K / BK;

#if HAVE_TDM
  // -------- Tensor Data Mover path: double-buffered DMA tiles --------
  __shared__ __align__(16) u16 lA[2][BM * BK];   // 2 x 16 KB
  __shared__ __align__(16) u16 lB[2][BN * BK];   // 2 x  4 KB
  if (wid == 0) {
    tdm_load_2d((unsigned)(uintptr_t)&lA[0][0], A + (size_t)m0 * lda, BK, BM,
                (unsigned)K, (unsigned)BM, (unsigned long long)lda);
    tdm_load_2d((unsigned)(uintptr_t)&lB[0][0], W + (size_t)n0 * ldw, BK, BN,
                (unsigned)K, (unsigned)BN, (unsigned long long)ldw);
  }
  for (int kt = 0; kt < nk; ++kt) {
    const int cur = kt & 1;
    if (wid == 0) {
      if (kt + 1 < nk) {
        const int k1 = (kt + 1) * BK;
        tdm_load_2d((unsigned)(uintptr_t)&lA[cur ^ 1][0], A + (size_t)m0 * lda + k1, BK, BM,
                    (unsigned)K, (unsigned)BM, (unsigned long long)lda);
        tdm_load_2d((unsigned)(uintptr_t)&lB[cur ^ 1][0], W + (size_t)n0 * ldw + k1, BK, BN,
                    (unsigned)K, (unsigned)BN, (unsigned long long)ldw);
        __builtin_amdgcn_s_wait_tensorcnt((short)2);   // current tile pair complete (in-order)
      } else {
        __builtin_amdgcn_s_wait_tensorcnt((short)0);
      }
    }
    __syncthreads();   // tile `cur` visible to all waves

    v16bf afrag[4];
    const int kA = (lane >> 4) * 8;
    #pragma unroll
    for (int mi = 0; mi < 4; ++mi) {
      int r = waveM * 64 + mi * 16 + (lane & 15);
      u16x8 h0 = *reinterpret_cast<const u16x8*>(&lA[cur][r * BK + kA]);
      u16x8 h1 = *reinterpret_cast<const u16x8*>(&lA[cur][r * BK + kA + 16]);
      u16x16 f = __builtin_shufflevector(h0, h1, 0,1,2,3,4,5,6,7,8,9,10,11,12,13,14,15);
      afrag[mi] = __builtin_bit_cast(v16bf, f);
    }
    v16bf bfrag[2];
    const int kB = (lane >> 4) * 8;
    #pragma unroll
    for (int ni = 0; ni < 2; ++ni) {
      int rr = waveN * 32 + ni * 16 + (lane & 15);
      u16x8 h0 = *reinterpret_cast<const u16x8*>(&lB[cur][rr * BK + kB]);
      u16x8 h1 = *reinterpret_cast<const u16x8*>(&lB[cur][rr * BK + kB + 16]);
      u16x16 f = __builtin_shufflevector(h0, h1, 0,1,2,3,4,5,6,7,8,9,10,11,12,13,14,15);
      bfrag[ni] = __builtin_bit_cast(v16bf, f);
    }
    #pragma unroll
    for (int mi = 0; mi < 4; ++mi)
      #pragma unroll
      for (int ni = 0; ni < 2; ++ni)
        acc[mi][ni] = __builtin_amdgcn_wmma_f32_16x16x32_bf16(
            false, afrag[mi], false, bfrag[ni], (short)0, acc[mi][ni], false, false);
    __syncthreads();   // reads of `cur` done before TDM overwrites it at kt+2
  }
#else
  // -------- fallback: manual cooperative staging (round-1 path) --------
  __shared__ __align__(16) u16 lA[1][BM * BK];
  __shared__ __align__(16) u16 lB[1][BN * BK];
  for (int kt = 0; kt < nk; ++kt) {
    const int k0 = kt * BK;
    #pragma unroll
    for (int c = 0; c < 4; ++c) {
      int ch  = tid * 4 + c;
      int row = ch >> 2;
      int kc  = (ch & 3) * 8;
      u32x4 v = *reinterpret_cast<const u32x4*>(A + (size_t)(m0 + row) * lda + k0 + kc);
      *reinterpret_cast<u32x4*>(&lA[0][row * BK + kc]) = v;
    }
    {
      int row = tid >> 2;
      int kc  = (tid & 3) * 8;
      u32x4 v = *reinterpret_cast<const u32x4*>(W + (size_t)(n0 + row) * ldw + k0 + kc);
      *reinterpret_cast<u32x4*>(&lB[0][row * BK + kc]) = v;
    }
    if (kt + 1 < nk) {
      __builtin_prefetch(A + (size_t)(m0 + (tid >> 2)) * lda + k0 + BK, 0, 3);
      __builtin_prefetch(W + (size_t)(n0 + (tid >> 2)) * ldw + k0 + BK, 0, 3);
    }
    __syncthreads();
    v16bf afrag[4];
    const int kA = (lane >> 4) * 8;
    #pragma unroll
    for (int mi = 0; mi < 4; ++mi) {
      int r = waveM * 64 + mi * 16 + (lane & 15);
      u16x8 h0 = *reinterpret_cast<const u16x8*>(&lA[0][r * BK + kA]);
      u16x8 h1 = *reinterpret_cast<const u16x8*>(&lA[0][r * BK + kA + 16]);
      u16x16 f = __builtin_shufflevector(h0, h1, 0,1,2,3,4,5,6,7,8,9,10,11,12,13,14,15);
      afrag[mi] = __builtin_bit_cast(v16bf, f);
    }
    v16bf bfrag[2];
    const int kB = (lane >> 4) * 8;
    #pragma unroll
    for (int ni = 0; ni < 2; ++ni) {
      int rr = waveN * 32 + ni * 16 + (lane & 15);
      u16x8 h0 = *reinterpret_cast<const u16x8*>(&lB[0][rr * BK + kB]);
      u16x8 h1 = *reinterpret_cast<const u16x8*>(&lB[0][rr * BK + kB + 16]);
      u16x16 f = __builtin_shufflevector(h0, h1, 0,1,2,3,4,5,6,7,8,9,10,11,12,13,14,15);
      bfrag[ni] = __builtin_bit_cast(v16bf, f);
    }
    #pragma unroll
    for (int mi = 0; mi < 4; ++mi)
      #pragma unroll
      for (int ni = 0; ni < 2; ++ni)
        acc[mi][ni] = __builtin_amdgcn_wmma_f32_16x16x32_bf16(
            false, afrag[mi], false, bfrag[ni], (short)0, acc[mi][ni], false, false);
    __syncthreads();
  }
#endif

  // epilogue: C layout — VGPR r: row = r + 8*(lane>=16), col = lane&15
  #pragma unroll
  for (int mi = 0; mi < 4; ++mi) {
    #pragma unroll
    for (int ni = 0; ni < 2; ++ni) {
      int col = n0 + waveN * 32 + ni * 16 + (lane & 15);
      float bv = bias ? bias[col] : 0.f;
      #pragma unroll
      for (int r = 0; r < 8; ++r) {
        int row = m0 + waveM * 64 + mi * 16 + (lane >> 4) * 8 + r;
        float v = apply_act(acc[mi][ni][r] + bv, act);
        C[(size_t)row * N + col] = v;
        if (Cbf) Cbf[(size_t)row * N + col] = f2bf(v);
      }
    }
  }
}

// ---------------- causal depthwise conv (D_CONV=2) + bias + SiLU ----------------
__global__ void k_conv_silu(const float* __restrict__ xz, const float* __restrict__ cw,
                            const float* __restrict__ cb, float* __restrict__ xi,
                            u16* __restrict__ xibf) {
  int i = blockIdx.x * blockDim.x + threadIdx.x;
  if (i >= M_TOK * DINNER) return;
  int c  = i & (DINNER - 1);
  int bl = i >> 10;
  int l  = bl & (L_ - 1);
  float x1 = xz[(size_t)bl * (2 * DINNER) + c];
  float x0 = (l > 0) ? xz[(size_t)(bl - 1) * (2 * DINNER) + c] : 0.f;
  float v  = cw[c * 2 + 0] * x0 + cw[c * 2 + 1] * x1 + cb[c];
  v = v / (1.f + __expf(-v));                        // SiLU
  xi[i]   = v;
  xibf[i] = f2bf(v);
}

// ---------------- selective scan: one thread per (b, d_inner) channel ----------------
__launch_bounds__(256, 4)
__global__ void k_scan(const float* __restrict__ xi, const float* __restrict__ dt,
                       const float* __restrict__ xdbl, const float* __restrict__ xz,
                       const float* __restrict__ A_log, const float* __restrict__ Dskip,
                       u16* __restrict__ ybf) {
  __shared__ float bc[2][32];                        // B_t(16) | C_t(16), double-buffered
  const int tid = threadIdx.x;
  const int b   = blockIdx.x >> 2;                   // 8 batches x 4 channel groups
  const int d   = (blockIdx.x & 3) * 256 + tid;
  float A[DSTATE], h[DSTATE];
  #pragma unroll
  for (int s = 0; s < DSTATE; ++s) { A[s] = -__expf(A_log[d * DSTATE + s]); h[s] = 0.f; }
  const float Dv = Dskip[d];

  for (int l = 0; l < L_; ++l) {
    const size_t bl = (size_t)b * L_ + l;
    if (tid < 32) bc[l & 1][tid] = xdbl[bl * 64 + 32 + tid];
    __syncthreads();
    const float* Bt = bc[l & 1];
    const float* Ct = bc[l & 1] + 16;
    float u   = xi[bl * DINNER + d];
    float dtv = dt[bl * DINNER + d];
    float du  = dtv * u;
    float y   = 0.f;
    #pragma unroll
    for (int s = 0; s < DSTATE; ++s) {
      h[s] = __expf(dtv * A[s]) * h[s] + du * Bt[s];
      y   += h[s] * Ct[s];
    }
    float z  = xz[bl * (2 * DINNER) + DINNER + d];
    float sz = z / (1.f + __expf(-z));
    ybf[bl * DINNER + d] = f2bf((y + u * Dv) * sz);
  }
}

// ---------------- x + fwd + flip(bwd) -> LayerNorm (f32 + bf16 out) ----------------
__global__ void k_add3_ln(const float* __restrict__ x, const float* __restrict__ of,
                          const float* __restrict__ ob, const float* __restrict__ g,
                          const float* __restrict__ be, float* __restrict__ out,
                          u16* __restrict__ outbf) {
  __shared__ float red[256];
  __shared__ float row[DMODEL];
  const int bl = blockIdx.x;
  const int l  = bl & (L_ - 1);
  const int b  = bl >> 11;
  const size_t flip = ((size_t)b * L_ + (L_ - 1 - l)) * DMODEL;
  const int tid = threadIdx.x;
  float local = 0.f;
  for (int i = tid; i < DMODEL; i += 256) {
    float v = x[(size_t)bl * DMODEL + i] + of[(size_t)bl * DMODEL + i] + ob[flip + i];
    row[i] = v; local += v;
  }
  red[tid] = local; __syncthreads();
  for (int s = 128; s > 0; s >>= 1) { if (tid < s) red[tid] += red[tid + s]; __syncthreads(); }
  float mean = red[0] * (1.f / DMODEL); __syncthreads();
  local = 0.f;
  for (int i = tid; i < DMODEL; i += 256) { float dv = row[i] - mean; local += dv * dv; }
  red[tid] = local; __syncthreads();
  for (int s = 128; s > 0; s >>= 1) { if (tid < s) red[tid] += red[tid + s]; __syncthreads(); }
  float rstd = rsqrtf(red[0] * (1.f / DMODEL) + 1e-5f);
  for (int i = tid; i < DMODEL; i += 256) {
    float v = (row[i] - mean) * rstd * g[i] + be[i];
    out[(size_t)bl * DMODEL + i] = v;
    outbf[(size_t)bl * DMODEL + i] = f2bf(v);
  }
}

// ---------------- (a + b) -> LayerNorm -> final f32 output ----------------
__global__ void k_add2_ln(const float* __restrict__ a, const float* __restrict__ c,
                          const float* __restrict__ g, const float* __restrict__ be,
                          float* __restrict__ out) {
  __shared__ float red[256];
  __shared__ float row[DMODEL];
  const int bl = blockIdx.x;
  const int tid = threadIdx.x;
  float local = 0.f;
  for (int i = tid; i < DMODEL; i += 256) {
    float v = a[(size_t)bl * DMODEL + i] + c[(size_t)bl * DMODEL + i];
    row[i] = v; local += v;
  }
  red[tid] = local; __syncthreads();
  for (int s = 128; s > 0; s >>= 1) { if (tid < s) red[tid] += red[tid + s]; __syncthreads(); }
  float mean = red[0] * (1.f / DMODEL); __syncthreads();
  local = 0.f;
  for (int i = tid; i < DMODEL; i += 256) { float dv = row[i] - mean; local += dv * dv; }
  red[tid] = local; __syncthreads();
  for (int s = 128; s > 0; s >>= 1) { if (tid < s) red[tid] += red[tid + s]; __syncthreads(); }
  float rstd = rsqrtf(red[0] * (1.f / DMODEL) + 1e-5f);
  for (int i = tid; i < DMODEL; i += 256)
    out[(size_t)bl * DMODEL + i] = (row[i] - mean) * rstd * g[i] + be[i];
}

// ---------------- host orchestration ----------------
extern "C" void kernel_launch(void* const* d_in, const int* in_sizes, int n_in,
                              void* d_out, int out_size, void* d_ws, size_t ws_size,
                              hipStream_t stream) {
  (void)in_sizes; (void)n_in; (void)out_size; (void)ws_size;
  const float* x = (const float*)d_in[0];
  const float* p_in[2]   = {(const float*)d_in[1],  (const float*)d_in[10]};
  const float* p_cw[2]   = {(const float*)d_in[2],  (const float*)d_in[11]};
  const float* p_cb[2]   = {(const float*)d_in[3],  (const float*)d_in[12]};
  const float* p_xp[2]   = {(const float*)d_in[4],  (const float*)d_in[13]};
  const float* p_dtw[2]  = {(const float*)d_in[5],  (const float*)d_in[14]};
  const float* p_dtb[2]  = {(const float*)d_in[6],  (const float*)d_in[15]};
  const float* p_alog[2] = {(const float*)d_in[7],  (const float*)d_in[16]};
  const float* p_dsk[2]  = {(const float*)d_in[8],  (const float*)d_in[17]};
  const float* p_op[2]   = {(const float*)d_in[9],  (const float*)d_in[18]};
  const float* w1  = (const float*)d_in[19];
  const float* b1  = (const float*)d_in[20];
  const float* w2  = (const float*)d_in[21];
  const float* b2  = (const float*)d_in[22];
  const float* l1g = (const float*)d_in[23];
  const float* l1b = (const float*)d_in[24];
  const float* l2g = (const float*)d_in[25];
  const float* l2b = (const float*)d_in[26];

  // bump allocator over d_ws
  char* base = (char*)d_ws;
  size_t off = 0;
  auto alloc = [&](size_t bytes) -> void* {
    void* r = base + off;
    off = (off + bytes + 255) & ~(size_t)255;
    return r;
  };
  u16* xbf      = (u16*)alloc((size_t)M_TOK * DMODEL * 2);
  u16* xflipbf  = (u16*)alloc((size_t)M_TOK * DMODEL * 2);
  u16* wbf_in[2]  = {(u16*)alloc(2048 * 512 * 2), (u16*)alloc(2048 * 512 * 2)};
  u16* wbf_xp[2]  = {(u16*)alloc(64 * 1024 * 2),  (u16*)alloc(64 * 1024 * 2)};
  u16* wbf_dtw[2] = {(u16*)alloc(1024 * 32 * 2),  (u16*)alloc(1024 * 32 * 2)};
  u16* wbf_op[2]  = {(u16*)alloc(512 * 1024 * 2), (u16*)alloc(512 * 1024 * 2)};
  u16* wbf_1      = (u16*)alloc(2048 * 512 * 2);
  u16* wbf_2      = (u16*)alloc(512 * 2048 * 2);
  float* xz   = (float*)alloc((size_t)M_TOK * 2048 * 4);  // reused later as FFN-h f32
  float* xi   = (float*)alloc((size_t)M_TOK * 1024 * 4);  // reused later as x_ffn f32
  u16*  xibf  = (u16*) alloc((size_t)M_TOK * 1024 * 2);
  float* xdbl = (float*)alloc((size_t)M_TOK * 64 * 4);
  u16*  xdblbf= (u16*) alloc((size_t)M_TOK * 64 * 2);
  float* dtb  = (float*)alloc((size_t)M_TOK * 1024 * 4);  // reused later as FFN-h bf16 (same bytes)
  u16*  ybf   = (u16*) alloc((size_t)M_TOK * 1024 * 2);
  float* odir[2] = {(float*)alloc((size_t)M_TOK * 512 * 4),
                    (float*)alloc((size_t)M_TOK * 512 * 4)};
  float* ln1f = (float*)alloc((size_t)M_TOK * 512 * 4);
  u16*  ln1bf = (u16*) alloc((size_t)M_TOK * 512 * 2);

  auto cast = [&](const float* s, u16* d, int n) {
    k_cast_bf16<<<(n + 255) / 256, 256, 0, stream>>>(s, d, n);
  };
  auto gemm = [&](const u16* A, int lda, const u16* W, int ldw, float* C, u16* Cbf,
                  const float* bias, int N, int K, int act) {
    dim3 g(N / BN, M_TOK / BM);
    k_gemm_bf16_wmma<<<g, 256, 0, stream>>>(A, lda, W, ldw, C, Cbf, bias, N, K, act);
  };

  // weight casts to bf16
  for (int d = 0; d < 2; ++d) {
    cast(p_in[d],  wbf_in[d],  2048 * 512);
    cast(p_xp[d],  wbf_xp[d],  64 * 1024);
    cast(p_dtw[d], wbf_dtw[d], 1024 * 32);
    cast(p_op[d],  wbf_op[d],  512 * 1024);
  }
  cast(w1, wbf_1, 2048 * 512);
  cast(w2, wbf_2, 512 * 2048);
  // activation casts (fwd: as-is, bwd: flipped along L)
  cast(x, xbf, M_TOK * DMODEL);
  k_cast_bf16_flipL<<<(M_TOK * DMODEL + 255) / 256, 256, 0, stream>>>(
      x, xflipbf, L_, DMODEL, M_TOK * DMODEL);

  const u16* xin[2] = {xbf, xflipbf};
  for (int d = 0; d < 2; ++d) {
    // in_proj: (M,512) x (2048,512)^T -> xz (M,2048)
    gemm(xin[d], DMODEL, wbf_in[d], DMODEL, xz, nullptr, nullptr, 2 * DINNER, DMODEL, ACT_NONE);
    // causal depthwise conv + SiLU on first half of xz
    k_conv_silu<<<(M_TOK * DINNER + 255) / 256, 256, 0, stream>>>(
        xz, p_cw[d], p_cb[d], xi, xibf);
    // x_proj: (M,1024) x (64,1024)^T -> x_dbl (M,64)
    gemm(xibf, DINNER, wbf_xp[d], DINNER, xdbl, xdblbf, nullptr, 64, DINNER, ACT_NONE);
    // dt = softplus( x_dbl[:, :32] x dt_w^T + dt_b ) -> (M,1024)
    gemm(xdblbf, 64, wbf_dtw[d], DTRANK, dtb, nullptr, p_dtb[d], DINNER, DTRANK, ACT_SOFTPLUS);
    // selective scan (+ Dskip skip, * silu(z)) -> y bf16 (M,1024)
    k_scan<<<B_ * 4, 256, 0, stream>>>(xi, dtb, xdbl, xz, p_alog[d], p_dsk[d], ybf);
    // out_proj: (M,1024) x (512,1024)^T -> direction output (M,512)
    gemm(ybf, DINNER, wbf_op[d], DINNER, odir[d], nullptr, nullptr, DMODEL, DINNER, ACT_NONE);
  }

  // x + fwd + flip(bwd) -> LN1
  k_add3_ln<<<M_TOK, 256, 0, stream>>>(x, odir[0], odir[1], l1g, l1b, ln1f, ln1bf);

  // FFN: gelu(ln1 x w1^T + b1) x w2^T + b2   (h aliases xz/dtb buffers)
  float* hF32 = xz;
  u16*   hBf  = (u16*)dtb;   // M*1024*4 bytes == M*2048*2 bytes
  gemm(ln1bf, DMODEL, wbf_1, DMODEL, hF32, hBf, b1, DFF, DMODEL, ACT_GELU);
  float* xffn = xi;          // M*512*4 <= M*1024*4
  gemm(hBf, DFF, wbf_2, DFF, xffn, nullptr, b2, DMODEL, DFF, ACT_NONE);

  // LN2 -> final output
  k_add2_ln<<<M_TOK, 256, 0, stream>>>(ln1f, xffn, l2g, l2b, (float*)d_out);
}
